// AttentionBlock_89137751261838
// MI455X (gfx1250) — compile-verified, matrix operand order checked by reading, not run on hardware
//
#include <hip/hip_runtime.h>
#include <hip/hip_bf16.h>
#include <math.h>

typedef _Float16 v16h __attribute__((ext_vector_type(16)));
typedef _Float16 v8h  __attribute__((ext_vector_type(8)));
typedef float    v8f  __attribute__((ext_vector_type(8)));

#define Bsz 8
#define Cdim 256
#define Ldim 2048

#if defined(__HIP_DEVICE_COMPILE__) && __has_builtin(__builtin_amdgcn_tensor_load_to_lds)
#define USE_TDM 1
#else
#define USE_TDM 0
#endif

// ---------------------------------------------------------------------------
// WMMA fragment helpers (ISA 7.12.2, 16-bit A/B layout)
// ---------------------------------------------------------------------------
__device__ __forceinline__ int frag_k(int lane, int j) {
    return 2 * j + ((j >= 4) ? 8 : 0) + ((lane >= 16) ? 8 : 0);
}

// Load a 16x32 f16 fragment from an LDS tile stored row-major [row][k].
__device__ __forceinline__ v16h load_frag(const _Float16* base, int row,
                                          int stride_h, int k0, int lane) {
    union { v16h v; unsigned u[8]; } f;
    const _Float16* p = base + (size_t)row * stride_h + k0;
#pragma unroll
    for (int j = 0; j < 8; ++j)
        f.u[j] = *(const unsigned*)(p + frag_k(lane, j));
    return f.v;
}

// ---------------------------------------------------------------------------
// Tensor Data Mover: 2D f16 tile (global, row stride) -> LDS (padded rows).
// pad_interval_code: pad after (2<<code) dwords; pad_amount_code: (code+1) dwords.
// ---------------------------------------------------------------------------
#if USE_TDM
typedef unsigned int u32x4 __attribute__((ext_vector_type(4)));
typedef int          i32x8 __attribute__((ext_vector_type(8)));
typedef int          i32x4 __attribute__((ext_vector_type(4)));

__device__ __forceinline__ void tdm_load_2d_f16(
    void* lds_dst, const _Float16* gsrc, unsigned tile_w, unsigned tile_h,
    unsigned row_stride_elems, unsigned pad_interval_code,
    unsigned pad_amount_code) {
    const unsigned long long ga = (unsigned long long)gsrc;
    const unsigned lds = (unsigned)(unsigned long long)lds_dst; // LDS offset = addr[31:0]
    u32x4 g0;
    g0.x = 1u;                                    // count=1 user descriptor
    g0.y = lds;                                   // D#.lds_addr
    g0.z = (unsigned)(ga & 0xFFFFFFFFull);        // global_addr[31:0]
    g0.w = (unsigned)((ga >> 32) & 0x01FFFFFFull) | (2u << 30); // [56:32]|type=2
    i32x8 g1;
    g1[0] = (int)((1u << 16)                      // data_size = 2 bytes
                  | (1u << 20)                    // pad_enable
                  | (pad_interval_code << 22) | (pad_amount_code << 25));
    g1[1] = (int)((tile_w & 0xFFFFu) << 16);      // tensor_dim0[15:0]
    g1[2] = (int)((tile_w >> 16) | ((tile_h & 0xFFFFu) << 16)); // dim0 hi | dim1 lo
    g1[3] = (int)((tile_h >> 16) | (tile_w << 16));             // dim1 hi | tile_dim0
    g1[4] = (int)(tile_h & 0xFFFFu);              // tile_dim1 (tile_dim2 = 0)
    g1[5] = (int)row_stride_elems;                // tensor_dim0_stride[31:0]
    g1[6] = 0;
    g1[7] = 0;
    const i32x4 z4 = {0, 0, 0, 0};
    const i32x8 z8 = {0, 0, 0, 0, 0, 0, 0, 0};
    __builtin_amdgcn_tensor_load_to_lds(g0, g1, z4, z4, z8, 0);
}
#define TDM_WAIT() __builtin_amdgcn_s_wait_tensorcnt(0)
#endif

// ---------------------------------------------------------------------------
// Kernel 1: conv1x1 projection with f16 output.
//   transposed==0 : outh[b][o][l]        (used for V)
//   transposed==1 : outh[(b*L)+l][o]     (used for Q, K -> TDM-friendly)
// grid = (L/128, C/64, B), block = 256 (8 waves)
// ---------------------------------------------------------------------------
__global__ __launch_bounds__(256) void qkv_proj_kernel(
    const float* __restrict__ x, const float* __restrict__ W,
    const float* __restrict__ bias, _Float16* __restrict__ outh,
    int transposed) {
    const int lBase = blockIdx.x * 128;
    const int oBase = blockIdx.y * 64;
    const int b     = blockIdx.z;

    __shared__ _Float16 lW[64][264];   // [o'][c]   A operand (M x K)
    __shared__ _Float16 lX[128][40];   // [l'][c']  B operand tile ([N][K])

    const int tid  = threadIdx.x;
    const int wave = tid >> 5;
    const int lane = tid & 31;
    const int mIdx = wave & 3;
    const int nBase = (wave >> 2) * 64;

    {   // preload W panel 64x256, f32 -> f16
        const int row = tid >> 2;
        const int c0  = (tid & 3) << 6;
        const float* wp = W + (size_t)(oBase + row) * Cdim + c0;
#pragma unroll 8
        for (int i = 0; i < 64; ++i) lW[row][c0 + i] = (_Float16)wp[i];
    }

    v8f acc[4] = {};
    for (int kk = 0; kk < 8; ++kk) {
        __syncthreads();
        {   // stage X chunk transposed into LDS (f32->f16 conversion in VALU)
            const int cc = tid >> 3;
            const int l0 = (tid & 7) << 4;
            const float* xp = x + ((size_t)b * Cdim + (kk * 32 + cc)) * Ldim
                              + lBase + l0;
#pragma unroll
            for (int i = 0; i < 16; ++i) lX[l0 + i][cc] = (_Float16)xp[i];
        }
        __syncthreads();

        v16h a = load_frag(&lW[0][0], mIdx * 16 + (lane & 15), 264, kk * 32, lane);
#pragma unroll
        for (int t = 0; t < 4; ++t) {
            v16h bf = load_frag(&lX[0][0], nBase + t * 16 + (lane & 15), 40, 0, lane);
            acc[t] = __builtin_amdgcn_wmma_f32_16x16x32_f16(
                false, a, false, bf, (short)0, acc[t], false, false);
        }
    }

    // C/D layout: element (m = r + 8*hi, n = lane&15)
    const int hi   = (lane >> 4) * 8;
    const int ncol = lane & 15;
    if (transposed) {
        // out[(b*L + l)][o]; per lane r=0..7 are 8 consecutive o -> one 16B store
#pragma unroll
        for (int t = 0; t < 4; ++t) {
            const int l  = lBase + nBase + t * 16 + ncol;
            const int o0 = oBase + mIdx * 16 + hi;
            v8h st;
#pragma unroll
            for (int r = 0; r < 8; ++r)
                st[r] = (_Float16)(acc[t][r] + bias[o0 + r]);
            *(v8h*)(outh + ((size_t)(b * Ldim + l)) * Cdim + o0) = st;
        }
    } else {
#pragma unroll
        for (int t = 0; t < 4; ++t) {
            const int l = lBase + nBase + t * 16 + ncol;
#pragma unroll
            for (int r = 0; r < 8; ++r) {
                const int o = oBase + mIdx * 16 + r + hi;
                outh[((size_t)b * Cdim + o) * Ldim + l] =
                    (_Float16)(acc[t][r] + bias[o]);
            }
        }
    }
}

// ---------------------------------------------------------------------------
// Kernel 2: flash attention, 32 query rows per workgroup.
//   Qt, Kt: f16 [B*L][C] (transposed), Vh: f16 [B][C][L]
// grid = (L/32, B), block = 256 (8 waves)
// ---------------------------------------------------------------------------
__global__ __launch_bounds__(256) void attn_kernel(
    const float* __restrict__ x, const _Float16* __restrict__ Qt,
    const _Float16* __restrict__ Kt, const _Float16* __restrict__ Vh,
    float* __restrict__ out) {
    const int lBase = blockIdx.x * 32;
    const int b     = blockIdx.y;

    __shared__ _Float16 lQ[32][264];   // [l'][c]   A operand for scores
    __shared__ _Float16 lKt[64][136];  // [m'][c']  B operand chunk (c-chunk=128)
    __shared__ float    lS[32][72];    // raw scores tile
    __shared__ _Float16 lP[32][72];    // exp(S - max) tile (A operand)
    __shared__ _Float16 lVt[64][72];   // [c'][m']  B operand chunk
    __shared__ float    run_max[32], run_sum[32], corr[32];

    const int tid  = threadIdx.x;
    const int wave = tid >> 5;
    const int lane = tid & 31;
    const int lIdx = wave & 1;   // query-row 16-tile
    const int nSel = wave >> 1;  // 0..3

    // ---- load Q tile [32 x 256] (row stride C) ------------------------
#if USE_TDM
    if (wave == 0) {
        tdm_load_2d_f16(&lQ[0][0], Qt + (size_t)(b * Ldim + lBase) * Cdim,
                        256, 32, Cdim, /*2<<6=128 dw*/ 6, /*4 dw*/ 3); // stride 264
        TDM_WAIT();
    }
#else
    {
        const int l  = tid >> 3;
        const int c0 = (tid & 7) << 5;
        const _Float16* qp = Qt + (size_t)(b * Ldim + lBase + l) * Cdim + c0;
#pragma unroll 8
        for (int i = 0; i < 32; ++i) lQ[l][c0 + i] = qp[i];
    }
#endif
    if (tid < 32) { run_max[tid] = -1e30f; run_sum[tid] = 0.0f; }

    v8f oacc[4] = {};
    const int hi   = (lane >> 4) * 8;
    const int ncol = lane & 15;

    for (int mb = 0; mb < Ldim / 64; ++mb) {
        const int mBase = mb * 64;

        // ---- scores S = Q K^T for this 32x64 tile ----------------------
        v8f sacc = {};
        for (int kk = 0; kk < 2; ++kk) {
            __syncthreads();
#if USE_TDM
            if (wave == 0) {
                tdm_load_2d_f16(&lKt[0][0],
                                Kt + (size_t)(b * Ldim + mBase) * Cdim + kk * 128,
                                128, 64, Cdim, /*2<<5=64 dw*/ 5, 3); // stride 136
                TDM_WAIT();
            }
#else
            {
                const int m  = tid >> 2;
                const int c0 = (tid & 3) << 5;
                const _Float16* kp =
                    Kt + (size_t)(b * Ldim + mBase + m) * Cdim + kk * 128 + c0;
#pragma unroll 8
                for (int i = 0; i < 32; ++i) lKt[m][c0 + i] = kp[i];
            }
#endif
            __syncthreads();
#pragma unroll
            for (int ks = 0; ks < 4; ++ks) {
                v16h a  = load_frag(&lQ[0][0], lIdx * 16 + (lane & 15), 264,
                                    kk * 128 + ks * 32, lane);
                v16h bf = load_frag(&lKt[0][0], nSel * 16 + (lane & 15), 136,
                                    ks * 32, lane);
                sacc = __builtin_amdgcn_wmma_f32_16x16x32_f16(
                    false, a, false, bf, (short)0, sacc, false, false);
            }
        }
#pragma unroll
        for (int r = 0; r < 8; ++r)
            lS[lIdx * 16 + r + hi][nSel * 16 + ncol] = sacc[r] * 0.0625f;
        __syncthreads();

        // ---- online softmax (one thread per query row) -----------------
        if (tid < 32) {
            const float mx = run_max[tid];
            float nm = mx;
#pragma unroll 8
            for (int j = 0; j < 64; ++j) nm = fmaxf(nm, lS[tid][j]);
            const float cr = __expf(mx - nm);
            float s = 0.0f;
#pragma unroll 8
            for (int j = 0; j < 64; ++j) {
                const float e = __expf(lS[tid][j] - nm);
                lP[tid][j] = (_Float16)e;
                s += e;
            }
            run_sum[tid] = run_sum[tid] * cr + s;
            run_max[tid] = nm;
            corr[tid]    = cr;
        }
        __syncthreads();

#pragma unroll
        for (int r = 0; r < 8; ++r) {
            const float f = corr[lIdx * 16 + r + hi];
#pragma unroll
            for (int t = 0; t < 4; ++t) oacc[t][r] *= f;
        }

        // ---- context O += P * V^T, streaming V in 64-channel chunks ----
        for (int j = 0; j < 4; ++j) {
            __syncthreads();
#if USE_TDM
            if (wave == 0) {
                tdm_load_2d_f16(&lVt[0][0],
                                Vh + ((size_t)b * Cdim + j * 64) * Ldim + mBase,
                                64, 64, Ldim, /*2<<4=32 dw*/ 4, 3); // stride 72
                TDM_WAIT();
            }
#else
            {
                const int cc = tid >> 2;
                const int m0 = (tid & 3) << 4;
                const _Float16* vp =
                    Vh + ((size_t)b * Cdim + j * 64 + cc) * Ldim + mBase + m0;
#pragma unroll
                for (int i = 0; i < 16; ++i) lVt[cc][m0 + i] = vp[i];
            }
#endif
            __syncthreads();
#pragma unroll
            for (int ks = 0; ks < 2; ++ks) {
                v16h a  = load_frag(&lP[0][0],  lIdx * 16 + (lane & 15), 72,
                                    ks * 32, lane);
                v16h bf = load_frag(&lVt[0][0], nSel * 16 + (lane & 15), 72,
                                    ks * 32, lane);
                oacc[j] = __builtin_amdgcn_wmma_f32_16x16x32_f16(
                    false, a, false, bf, (short)0, oacc[j], false, false);
            }
        }
    }
    __syncthreads();

    // ---- finalize: out[b,c,l] = O[l,c]/rowsum[l] + x[b,c,l] ----------------
#pragma unroll
    for (int r = 0; r < 8; ++r) {
        const int l     = lBase + lIdx * 16 + r + hi;
        const float inv = 1.0f / run_sum[lIdx * 16 + r + hi];
#pragma unroll
        for (int t = 0; t < 4; ++t) {
            const int c = t * 64 + nSel * 16 + ncol;
            const size_t idx = ((size_t)b * Cdim + c) * Ldim + l;
            out[idx] = oacc[t][r] * inv + x[idx];
        }
    }
}

// ---------------------------------------------------------------------------
extern "C" void kernel_launch(void* const* d_in, const int* in_sizes, int n_in,
                              void* d_out, int out_size, void* d_ws, size_t ws_size,
                              hipStream_t stream) {
    const float* x  = (const float*)d_in[0];
    const float* Wq = (const float*)d_in[1];
    const float* bq = (const float*)d_in[2];
    const float* Wk = (const float*)d_in[3];
    const float* bk = (const float*)d_in[4];
    const float* Wv = (const float*)d_in[5];
    const float* bv = (const float*)d_in[6];
    float* out = (float*)d_out;

    const size_t elems = (size_t)Bsz * Cdim * Ldim;
    _Float16* Qt = (_Float16*)d_ws;       // [B*L][C]
    _Float16* Kt = Qt + elems;            // [B*L][C]
    _Float16* Vh = Kt + elems;            // [B][C][L]

    dim3 gProj(Ldim / 128, Cdim / 64, Bsz);
    qkv_proj_kernel<<<gProj, 256, 0, stream>>>(x, Wq, bq, Qt, 1);
    qkv_proj_kernel<<<gProj, 256, 0, stream>>>(x, Wk, bk, Kt, 1);
    qkv_proj_kernel<<<gProj, 256, 0, stream>>>(x, Wv, bv, Vh, 0);

    dim3 gAttn(Ldim / 32, Bsz);
    attn_kernel<<<gAttn, 256, 0, stream>>>(x, Qt, Kt, Vh, out);
}
